// MultiHeadAttention_33054068310067
// MI455X (gfx1250) — matrix operand order repaired and checked by reading, not compiled, not run on hardware
//
#include <hip/hip_runtime.h>
#include <hip/hip_bf16.h>

typedef __attribute__((ext_vector_type(16))) _Float16 v16h;
typedef __attribute__((ext_vector_type(8)))  _Float16 v8h;
typedef __attribute__((ext_vector_type(8)))  float    v8f;
typedef __attribute__((ext_vector_type(4)))  int      v4i;

#define D_MODEL 1024
#define N_HEADS 16
#define D_K     64
#define SEQ     2048
#define BATCH   4
#define SCALE   0.125f
#define NEGINF  (-1e30f)

#define WMMA_F16(a, b, c) \
  __builtin_amdgcn_wmma_f32_16x16x32_f16(false, (a), false, (b), (short)0, (c), false, false)

#if __has_builtin(__builtin_amdgcn_global_load_async_to_lds_b128)
#define HAVE_ASYNC_LDS 1
#endif

// ---- async global -> LDS helpers (gfx1250 ASYNCcnt path) ----------------
__device__ __forceinline__ void copy16_to_lds(const _Float16* g, _Float16* l) {
#ifdef HAVE_ASYNC_LDS
  __builtin_amdgcn_global_load_async_to_lds_b128(
      (__attribute__((address_space(1))) v4i*)(g),
      (__attribute__((address_space(3))) v4i*)(l),
      0, 0);
#else
  *(v8h*)l = *(const v8h*)g;
#endif
}

__device__ __forceinline__ void wait_async_done() {
#ifdef HAVE_ASYNC_LDS
#if __has_builtin(__builtin_amdgcn_s_wait_asynccnt)
  __builtin_amdgcn_s_wait_asynccnt(0);
#else
  asm volatile("s_wait_asynccnt 0x0" ::: "memory");
#endif
#endif
}

// Stage a 64-row x 32-half tile (row stride `gld` halves) into LDS [row][32].
// 256 chunks of 16B, 128 threads -> 2 chunks per thread.
__device__ __forceinline__ void stage_tile32(const _Float16* gsrc, size_t gld,
                                             _Float16* lds, int tid) {
#pragma unroll
  for (int i = 0; i < 2; ++i) {
    const int c = tid + 128 * i;        // 0..255
    const int row = c >> 2;             // 0..63
    const int ko  = (c & 3) * 8;        // 0,8,16,24
    copy16_to_lds(gsrc + (size_t)row * gld + ko, lds + row * 32 + ko);
  }
}
// Stage a 32-row x 64-half tile (row stride gld) into LDS [row][64].
__device__ __forceinline__ void stage_tile64(const _Float16* gsrc, size_t gld,
                                             _Float16* lds, int tid) {
#pragma unroll
  for (int i = 0; i < 2; ++i) {
    const int c = tid + 128 * i;        // 0..255
    const int row = c >> 3;             // 0..31
    const int ko  = (c & 7) * 8;        // 0..56
    copy16_to_lds(gsrc + (size_t)row * gld + ko, lds + row * 64 + ko);
  }
}

// ---- fragment loaders (ISA 7.12.2 layouts, wave32) ----------------------
__device__ __forceinline__ v16h load_a16(const _Float16* base, int m0, int k0, int ld) {
  const int lane = threadIdx.x & 31;
  const _Float16* rp = base + (size_t)(m0 + (lane & 15)) * ld + k0 + ((lane >> 4) << 3);
  union { v16h v; v8h h[2]; } u;
  u.h[0] = *(const v8h*)(rp);
  u.h[1] = *(const v8h*)(rp + 16);
  return u.v;
}
__device__ __forceinline__ v16h load_b16(const _Float16* base, int n0, int k0, int ld) {
  const int lane = threadIdx.x & 31;
  const _Float16* cp = base + (size_t)(n0 + (lane & 15)) * ld + k0 + ((lane >> 4) << 3);
  union { v16h v; v8h h[2]; } u;
  u.h[0] = *(const v8h*)(cp);
  u.h[1] = *(const v8h*)(cp + 16);
  return u.v;
}

// ---- kernel 0: transpose-convert x (B,D,S) f32 -> Xt (B*S, D) f16 -------
__global__ void __launch_bounds__(256) transpose_cvt(const float* __restrict__ x,
                                                     _Float16* __restrict__ Xt) {
  __shared__ float tile[32][33];
  const int b  = blockIdx.z;
  const int s0 = blockIdx.x * 32;
  const int d0 = blockIdx.y * 32;
  const int tx = threadIdx.x, ty = threadIdx.y;          // 32 x 8
  const float* xb = x + (size_t)b * D_MODEL * SEQ;
#pragma unroll
  for (int i = 0; i < 4; ++i)
    tile[ty + 8 * i][tx] = xb[(size_t)(d0 + ty + 8 * i) * SEQ + s0 + tx];
  __syncthreads();
  _Float16* xo = Xt + (size_t)b * SEQ * D_MODEL;
#pragma unroll
  for (int i = 0; i < 4; ++i)
    xo[(size_t)(s0 + ty + 8 * i) * D_MODEL + d0 + tx] = (_Float16)tile[tx][ty + 8 * i];
}

// ---- kernel 1: f32 -> f16 convert ---------------------------------------
__global__ void __launch_bounds__(256) cvt_f16(const float* __restrict__ src,
                                               _Float16* __restrict__ dst, int n) {
  int i = blockIdx.x * 256 + threadIdx.x;
  if (i < n) dst[i] = (_Float16)src[i];
}

// ---- kernel 2: WMMA GEMM  Y[s,o] = sum_d A[s,d]*W[o,d] + bias[o] --------
// Each wave owns a 32x64 output tile (8 WMMA per k-step); the 64x32 W tile is
// staged to LDS async + double buffered and shared by all 4 waves; A-fragment
// global loads are software-pipelined one k-step ahead.
// mode 0: Y -> (B,H,S,64) f16 ; mode 1: Y -> (B,H,64,S) f16 ; mode 2: (B,D,S) f32
__global__ void __launch_bounds__(128) gemm_wmma(const _Float16* __restrict__ A,
                                                 const _Float16* __restrict__ W,
                                                 const float* __restrict__ bias,
                                                 _Float16* __restrict__ outh,
                                                 float* __restrict__ outf, int mode) {
  __shared__ _Float16 Wt[2][64 * 32];
  const int tid  = threadIdx.x;
  const int lane = tid & 31;
  const int wave = tid >> 5;
  const int m0 = blockIdx.y * 128 + wave * 32;   // 32 rows per wave
  const int n0 = blockIdx.x * 64;
  const v8f zf = {0.f, 0.f, 0.f, 0.f, 0.f, 0.f, 0.f, 0.f};
  v8f acc[2][4] = {{zf, zf, zf, zf}, {zf, zf, zf, zf}};

  stage_tile32(W + (size_t)n0 * D_MODEL, D_MODEL, &Wt[0][0], tid);
  v16h a0 = load_a16(A, m0, 0, D_MODEL);         // pipelined A fragments
  v16h a1 = load_a16(A, m0 + 16, 0, D_MODEL);

  for (int k0 = 0; k0 < D_MODEL; k0 += 32) {
    const int cur = (k0 >> 5) & 1;
    wait_async_done();
    __syncthreads();                       // Wt[cur] visible to all waves
    v16h a0n = a0, a1n = a1;
    if (k0 + 32 < D_MODEL) {               // overlap next stage with compute
      stage_tile32(W + (size_t)n0 * D_MODEL + (k0 + 32), D_MODEL, &Wt[cur ^ 1][0], tid);
      a0n = load_a16(A, m0, k0 + 32, D_MODEL);
      a1n = load_a16(A, m0 + 16, k0 + 32, D_MODEL);
    }
#pragma unroll
    for (int j = 0; j < 4; ++j) {
      v16h bf = load_b16(&Wt[cur][0], 16 * j, 0, 32);   // ds_load fragments
      acc[0][j] = WMMA_F16(a0, bf, acc[0][j]);
      acc[1][j] = WMMA_F16(a1, bf, acc[1][j]);
    }
    a0 = a0n;
    a1 = a1n;
    __syncthreads();                       // done reading Wt[cur]
  }

  const int rhi = (lane >> 4) << 3;
#pragma unroll
  for (int mi = 0; mi < 2; ++mi) {
#pragma unroll
    for (int j = 0; j < 4; ++j) {
      const int o = n0 + 16 * j + (lane & 15);
      const float bv = bias[o];
#pragma unroll
      for (int r = 0; r < 8; ++r) {
        const int sg = m0 + mi * 16 + r + rhi;
        const float v = acc[mi][j][r] + bv;
        const int b = sg >> 11, s = sg & (SEQ - 1);
        if (mode == 0) {
          const int h = o >> 6, d = o & 63;
          outh[((size_t)(b * N_HEADS + h) * SEQ + s) * D_K + d] = (_Float16)v;
        } else if (mode == 1) {
          const int h = o >> 6, d = o & 63;
          outh[((size_t)(b * N_HEADS + h) * D_K + d) * SEQ + s] = (_Float16)v;
        } else {
          outf[(size_t)(b * D_MODEL + o) * SEQ + s] = v;
        }
      }
    }
  }
}

// ---- kernel 3: flash attention, K/V tiles staged async to LDS -----------
__global__ void __launch_bounds__(128) attn_wmma(const _Float16* __restrict__ Q,
                                                 const _Float16* __restrict__ K,
                                                 const _Float16* __restrict__ Vt,
                                                 const int* __restrict__ mask,
                                                 _Float16* __restrict__ ctx) {
  __shared__ _Float16 Kt[2][32 * 64];     // [key_local][d]
  __shared__ _Float16 Vl[2][64 * 32];     // [d'][key_local]
  __shared__ _Float16 Pbuf[4][16 * 32];
  const int tid  = threadIdx.x;
  const int lane = tid & 31;
  const int wave = tid >> 5;
  const int b = blockIdx.z, h = blockIdx.y;
  const int q0 = (blockIdx.x * 4 + wave) * 16;
  const _Float16* Qb = Q  + (size_t)(b * N_HEADS + h) * SEQ * D_K;
  const _Float16* Kb = K  + (size_t)(b * N_HEADS + h) * SEQ * D_K;
  const _Float16* Vb = Vt + (size_t)(b * N_HEADS + h) * D_K * SEQ;
  const int* mb = mask + b * SEQ;
  _Float16* Pw = &Pbuf[wave][0];

  const v16h qa0 = load_a16(Qb, q0, 0, D_K);
  const v16h qa1 = load_a16(Qb, q0, 32, D_K);

  const v8f zf = {0.f, 0.f, 0.f, 0.f, 0.f, 0.f, 0.f, 0.f};
  v8f acc[4] = {zf, zf, zf, zf};
  float mrow[8], lrow[8];
#pragma unroll
  for (int r = 0; r < 8; ++r) { mrow[r] = NEGINF; lrow[r] = 0.f; }

  // prologue stage of tile 0
  stage_tile64(Kb, D_K, &Kt[0][0], tid);          // 32 keys x 64 d
  stage_tile32(Vb, SEQ, &Vl[0][0], tid);          // 64 d'  x 32 keys

  const int rhi = (lane >> 4) << 3;
  for (int kt = 0; kt < SEQ; kt += 32) {
    const int cur = (kt >> 5) & 1;
    wait_async_done();
    __syncthreads();                               // K/V tile[cur] ready
    if (kt + 32 < SEQ) {                           // overlap next stage
      stage_tile64(Kb + (size_t)(kt + 32) * D_K, D_K, &Kt[cur ^ 1][0], tid);
      stage_tile32(Vb + (kt + 32), SEQ, &Vl[cur ^ 1][0], tid);
    }

    // scores S = Q * K^T  (16 x 32), dk = 64 -> two k-steps per key half
    v16h b00 = load_b16(&Kt[cur][0], 0,  0,  D_K);
    v16h b01 = load_b16(&Kt[cur][0], 0,  32, D_K);
    v16h b10 = load_b16(&Kt[cur][0], 16, 0,  D_K);
    v16h b11 = load_b16(&Kt[cur][0], 16, 32, D_K);
    v8f s0 = WMMA_F16(qa0, b00, zf); s0 = WMMA_F16(qa1, b01, s0);
    v8f s1 = WMMA_F16(qa0, b10, zf); s1 = WMMA_F16(qa1, b11, s1);

    const int mk0 = mb[kt + (lane & 15)];
    const int mk1 = mb[kt + 16 + (lane & 15)];
    float alpha[8];
#pragma unroll
    for (int r = 0; r < 8; ++r) {
      float x0 = mk0 ? s0[r] * SCALE : NEGINF;
      float x1 = mk1 ? s1[r] * SCALE : NEGINF;
      float mx = fmaxf(x0, x1);
#pragma unroll
      for (int off = 1; off < 16; off <<= 1)
        mx = fmaxf(mx, __shfl_xor(mx, off, 32));   // 16-lane row-max
      const float mnew = fmaxf(mrow[r], mx);
      const float al = __expf(mrow[r] - mnew);
      const float p0 = __expf(x0 - mnew);
      const float p1 = __expf(x1 - mnew);
      float rs = p0 + p1;
#pragma unroll
      for (int off = 1; off < 16; off <<= 1)
        rs += __shfl_xor(rs, off, 32);             // 16-lane row-sum
      lrow[r] = lrow[r] * al + rs;
      mrow[r] = mnew;
      alpha[r] = al;
      const int row = r + rhi;
      Pw[row * 32 + (lane & 15)]      = (_Float16)p0;
      Pw[row * 32 + (lane & 15) + 16] = (_Float16)p1;
    }
#pragma unroll
    for (int j = 0; j < 4; ++j)
#pragma unroll
      for (int r = 0; r < 8; ++r) acc[j][r] = acc[j][r] * alpha[r];

    // P back as A-fragment (same-wave LDS ordering via DScnt)
    v16h pa = load_a16(Pw, 0, 0, 32);
#pragma unroll
    for (int j = 0; j < 4; ++j) {
      v16h vb = load_b16(&Vl[cur][0], j * 16, 0, 32);
      acc[j] = WMMA_F16(pa, vb, acc[j]);
    }
    __syncthreads();                               // done reading tile[cur]
  }
  // epilogue: ctx[b][s][h*64+d] = acc / l
#pragma unroll
  for (int j = 0; j < 4; ++j) {
    const int o = h * D_K + j * 16 + (lane & 15);
#pragma unroll
    for (int r = 0; r < 8; ++r) {
      const int s = q0 + r + rhi;
      ctx[(size_t)(b * SEQ + s) * D_MODEL + o] = (_Float16)(acc[j][r] / lrow[r]);
    }
  }
}

extern "C" void kernel_launch(void* const* d_in, const int* in_sizes, int n_in,
                              void* d_out, int out_size, void* d_ws, size_t ws_size,
                              hipStream_t stream) {
  const float* x   = (const float*)d_in[0];
  const int*  mask = (const int*)d_in[1];
  const float* Wq = (const float*)d_in[2]; const float* bq = (const float*)d_in[3];
  const float* Wk = (const float*)d_in[4]; const float* bk = (const float*)d_in[5];
  const float* Wv = (const float*)d_in[6]; const float* bv = (const float*)d_in[7];
  const float* Wo = (const float*)d_in[8]; const float* bo = (const float*)d_in[9];

  char* ws = (char*)d_ws;
  const size_t BIG = (size_t)BATCH * SEQ * D_MODEL * sizeof(_Float16);   // 16.8 MB
  const size_t WSZ = (size_t)D_MODEL * D_MODEL * sizeof(_Float16);       // 2.1 MB
  _Float16* Xt  = (_Float16*)ws;  ws += BIG;
  _Float16* Qb  = (_Float16*)ws;  ws += BIG;
  _Float16* Kb  = (_Float16*)ws;  ws += BIG;
  _Float16* Vb  = (_Float16*)ws;  ws += BIG;
  _Float16* Ctx = (_Float16*)ws;  ws += BIG;
  _Float16* Wqh = (_Float16*)ws;  ws += WSZ;
  _Float16* Wkh = (_Float16*)ws;  ws += WSZ;
  _Float16* Wvh = (_Float16*)ws;  ws += WSZ;
  _Float16* Woh = (_Float16*)ws;  ws += WSZ;

  transpose_cvt<<<dim3(SEQ / 32, D_MODEL / 32, BATCH), dim3(32, 8), 0, stream>>>(x, Xt);
  const int NW = D_MODEL * D_MODEL;
  cvt_f16<<<(NW + 255) / 256, 256, 0, stream>>>(Wq, Wqh, NW);
  cvt_f16<<<(NW + 255) / 256, 256, 0, stream>>>(Wk, Wkh, NW);
  cvt_f16<<<(NW + 255) / 256, 256, 0, stream>>>(Wv, Wvh, NW);
  cvt_f16<<<(NW + 255) / 256, 256, 0, stream>>>(Wo, Woh, NW);

  dim3 ggrid(D_MODEL / 64, (BATCH * SEQ) / 128, 1);   // waves own 32 rows now
  gemm_wmma<<<ggrid, 128, 0, stream>>>(Xt, Wqh, bq, Qb, nullptr, 0);
  gemm_wmma<<<ggrid, 128, 0, stream>>>(Xt, Wkh, bk, Kb, nullptr, 0);
  gemm_wmma<<<ggrid, 128, 0, stream>>>(Xt, Wvh, bv, Vb, nullptr, 1);

  attn_wmma<<<dim3(SEQ / 64, N_HEADS, BATCH), 128, 0, stream>>>(Qb, Kb, Vb, mask, Ctx);

  gemm_wmma<<<ggrid, 128, 0, stream>>>(Ctx, Woh, bo, nullptr, (float*)d_out, 2);
}
